// InvertibleSASModelCore_3539053052720
// MI455X (gfx1250) — matrix operand order repaired and checked by reading, not compiled, not run on hardware
//
#include <hip/hip_runtime.h>
#include <hip/hip_bf16.h>
#include <math.h>

typedef __attribute__((ext_vector_type(16))) __bf16 v16bf;
typedef __attribute__((ext_vector_type(2)))  __bf16 v2bf;
typedef __attribute__((ext_vector_type(8)))  float  v8f;

#define NBLOCKS 5
#define ST_STRIDE 132   // f32 state row stride (128 cols + pad), 528B: 16B aligned, bank-spread
#define A64_STRIDE 72   // bf16 activation stride for 64-wide (144B)
#define A32_STRIDE 40   // bf16 activation stride for 32-wide (80B)
#define WK64_STRIDE 72  // bf16 W^T stride, K=64
#define WK32_STRIDE 40  // bf16 W^T stride, K=32

// ---- LDS carve-up (all offsets multiples of 16 bytes) ----
// Single per-sub-MLP weight buffer (staged 4x per block) -> 134.8KB total
// => 2 workgroups per WGP (269.6KB of 320KB), 16 waves/WGP for latency hiding.
constexpr int OFF_STATE = 0;
constexpr int OFF_H2    = OFF_STATE + 128 * ST_STRIDE * 4;     // 67584
constexpr int OFF_H1    = OFF_H2    + 128 * A64_STRIDE * 2;    // 86016
constexpr int OFF_XIN   = OFF_H1    + 128 * A32_STRIDE * 2;    // 96256
constexpr int OFF_W1T   = OFF_XIN   + 128 * A64_STRIDE * 2;    // 114688
constexpr int OFF_W2T   = OFF_W1T   + 64 * WK64_STRIDE * 2;    // 123904
constexpr int OFF_W3T   = OFF_W2T   + 32 * WK64_STRIDE * 2;    // 128512
constexpr int OFF_B1    = OFF_W3T   + 64 * WK32_STRIDE * 2;    // 133632
constexpr int OFF_B2    = OFF_B1    + 64 * 4;                  // 133888
constexpr int OFF_B3    = OFF_B2    + 32 * 4;                  // 134016
constexpr int OFF_PERM  = OFF_B3    + 64 * 4;                  // 134272
constexpr int SMEM_BYTES = OFF_PERM + 128 * 4;                 // 134784

union FragBF { v16bf v; uint4 q[2]; };
union FragF  { v8f   v; float f[8]; };

// Packed f32x2 -> bf16x2 using the hardware cvt when available.
static __device__ __forceinline__ unsigned pack_bf16x2(float lo, float hi) {
#if __has_builtin(__builtin_amdgcn_cvt_pk_bf16_f32)
  v2bf v = __builtin_amdgcn_cvt_pk_bf16_f32(lo, hi);
  return __builtin_bit_cast(unsigned, v);
#else
  v2bf v;
  v[0] = (__bf16)lo;
  v[1] = (__bf16)hi;
  return __builtin_bit_cast(unsigned, v);
#endif
}

// One packed cvt feeding two b16 stores (hi half via d16_hi store pattern).
static __device__ __forceinline__ void store_bf16_pair(__bf16* p0, __bf16* p1,
                                                       float v0, float v1) {
  unsigned pk = pack_bf16x2(v0, v1);
  *(unsigned short*)p0 = (unsigned short)pk;
  *(unsigned short*)p1 = (unsigned short)(pk >> 16);
}

// Minimax atan: ~1e-6 max error (far below the bf16 matmul noise floor).
// |x|<=1: odd poly (6 FMA); |x|>1: pi/2 - atan(1/x) via v_rcp_f32.
static __device__ __forceinline__ float fast_atan(float v) {
  float ax  = fabsf(v);
  bool  big = ax > 1.0f;
  float z   = big ? __builtin_amdgcn_rcpf(ax) : ax;
  float z2  = z * z;
  float p;
  p = fmaf(z2, -0.01172120f, 0.05265332f);
  p = fmaf(z2, p, -0.11643287f);
  p = fmaf(z2, p, 0.19354346f);
  p = fmaf(z2, p, -0.33262347f);
  p = fmaf(z2, p, 0.99997726f);
  p = z * p;
  float r = big ? (1.57079632679f - p) : p;
  return copysignf(r, v);
}

static __device__ __forceinline__ v8f wmma_bf16(v16bf a, v16bf b, v8f c) {
  return __builtin_amdgcn_wmma_f32_16x16x32_bf16(false, a, false, b, (short)0, c,
                                                 false, false);
}

// A fragment (16x32 bf16) from bf16 LDS buffer (two ds_load_b128 per lane).
// ISA layout: lanes 0-15: M=lane, K = kb+{0..7}, kb+{16..23};
//             lanes 16-31: M=lane-16, K = kb+{8..15}, kb+{24..31}.
static __device__ __forceinline__ v16bf a_frag_bf16(const __bf16* base, int stride,
                                                    int lane, int kb) {
  int m = lane & 15, half = lane >> 4;
  const __bf16* p = base + m * stride + kb + (half ? 8 : 0);
  FragBF u;
  u.q[0] = *(const uint4*)(p);
  u.q[1] = *(const uint4*)(p + 16);
  return u.v;
}

// B fragment (32x16 bf16) from LDS W^T ([N][K], N-major): lanes 0-15 K=kb+{0..7},kb+{16..23},
// lanes 16-31 K=kb+{8..15},kb+{24..31}. Two ds_load_b128 per lane.
static __device__ __forceinline__ v16bf b_frag(const __bf16* wT, int kstride,
                                               int lane, int ntile, int kb) {
  int n = lane & 15, half = lane >> 4;
  const __bf16* p = wT + (ntile * 16 + n) * kstride + kb + half * 8;
  FragBF u;
  u.q[0] = *(const uint4*)(p);
  u.q[1] = *(const uint4*)(p + 16);
  return u.v;
}

// 3-layer MLP over this wave's 16 rows; result (layer3 + bias) left in out[4].
// D-fragments: lane n = lane&15 -> col nt*16+n ; reg r -> row r + 8*(lane>>4).
static __device__ __forceinline__ void mlp_forward(
    const __bf16* xinRow, __bf16* h2Row, __bf16* h1Row,
    const __bf16* W1T, const __bf16* W2T, const __bf16* W3T,
    const float* b1, const float* b2, const float* b3,
    int lane, FragF out[4]) {
  const int n  = lane & 15;
  const int mh = (lane >> 4) * 8;

  // ---- layer 1: [16x64] @ [64x64] + b, ReLU -> h2 (bf16) ----
  {
    FragF acc[4];
#pragma unroll
    for (int nt = 0; nt < 4; ++nt)
#pragma unroll
      for (int r = 0; r < 8; ++r) acc[nt].f[r] = 0.0f;
#pragma unroll
    for (int k = 0; k < 2; ++k) {
      v16bf a = a_frag_bf16(xinRow, A64_STRIDE, lane, 32 * k);
#pragma unroll
      for (int nt = 0; nt < 4; ++nt) {
        v16bf b = b_frag(W1T, WK64_STRIDE, lane, nt, 32 * k);
        acc[nt].v = wmma_bf16(a, b, acc[nt].v);
      }
    }
#pragma unroll
    for (int nt = 0; nt < 4; ++nt) {
      float bias = b1[nt * 16 + n];
      __bf16* col = h2Row + mh * A64_STRIDE + nt * 16 + n;
#pragma unroll
      for (int r = 0; r < 8; r += 2) {
        float v0 = fmaxf(acc[nt].f[r] + bias, 0.0f);
        float v1 = fmaxf(acc[nt].f[r + 1] + bias, 0.0f);
        store_bf16_pair(col + r * A64_STRIDE, col + (r + 1) * A64_STRIDE, v0, v1);
      }
    }
  }
  // ---- layer 2: [16x64] @ [64x32] + b, ReLU -> h1 (bf16) ----
  {
    FragF acc[2];
#pragma unroll
    for (int nt = 0; nt < 2; ++nt)
#pragma unroll
      for (int r = 0; r < 8; ++r) acc[nt].f[r] = 0.0f;
#pragma unroll
    for (int k = 0; k < 2; ++k) {
      v16bf a = a_frag_bf16(h2Row, A64_STRIDE, lane, 32 * k);
#pragma unroll
      for (int nt = 0; nt < 2; ++nt) {
        v16bf b = b_frag(W2T, WK64_STRIDE, lane, nt, 32 * k);
        acc[nt].v = wmma_bf16(a, b, acc[nt].v);
      }
    }
#pragma unroll
    for (int nt = 0; nt < 2; ++nt) {
      float bias = b2[nt * 16 + n];
      __bf16* col = h1Row + mh * A32_STRIDE + nt * 16 + n;
#pragma unroll
      for (int r = 0; r < 8; r += 2) {
        float v0 = fmaxf(acc[nt].f[r] + bias, 0.0f);
        float v1 = fmaxf(acc[nt].f[r + 1] + bias, 0.0f);
        store_bf16_pair(col + r * A32_STRIDE, col + (r + 1) * A32_STRIDE, v0, v1);
      }
    }
  }
  // ---- layer 3: [16x32] @ [32x64] + b -> out (registers) ----
  {
#pragma unroll
    for (int nt = 0; nt < 4; ++nt)
#pragma unroll
      for (int r = 0; r < 8; ++r) out[nt].f[r] = 0.0f;
    v16bf a = a_frag_bf16(h1Row, A32_STRIDE, lane, 0);
#pragma unroll
    for (int nt = 0; nt < 4; ++nt) {
      v16bf b = b_frag(W3T, WK32_STRIDE, lane, nt, 0);
      out[nt].v = wmma_bf16(a, b, out[nt].v);
    }
#pragma unroll
    for (int nt = 0; nt < 4; ++nt) {
      float bias = b3[nt * 16 + n];
#pragma unroll
      for (int r = 0; r < 8; ++r) out[nt].f[r] += bias;
    }
  }
}

// y = exp(1.272*atan(s)) * state[xCol] + t ; write state[dstCol] (+ bf16 mirror).
static __device__ __forceinline__ void couple_epilogue(
    float* stRow, __bf16* mirRow, const FragF s[4], const FragF t[4],
    int lane, int xCol, int dstCol, bool writeMir) {
  const int n  = lane & 15;
  const int mh = (lane >> 4) * 8;
#pragma unroll
  for (int nt = 0; nt < 4; ++nt) {
    float y[8];
#pragma unroll
    for (int r = 0; r < 8; ++r) {
      int row = r + mh, col = nt * 16 + n;
      float es = __expf(1.272f * fast_atan(s[nt].f[r]));  // CLAMP*0.636 = 1.272
      y[r] = es * stRow[row * ST_STRIDE + xCol + col] + t[nt].f[r];
      stRow[row * ST_STRIDE + dstCol + col] = y[r];
    }
    if (writeMir) {
      __bf16* col = mirRow + mh * A64_STRIDE + nt * 16 + n;
#pragma unroll
      for (int r = 0; r < 8; r += 2)
        store_bf16_pair(col + r * A64_STRIDE, col + (r + 1) * A64_STRIDE,
                        y[r], y[r + 1]);
    }
  }
}

// Stage one sub-MLP's weights (transposed, bf16) + biases into LDS.
static __device__ __forceinline__ void stage_weights(
    int tid, int i, int j,
    const float* __restrict__ W1g, const float* __restrict__ W2g,
    const float* __restrict__ W3g, const float* __restrict__ b1g,
    const float* __restrict__ b2g, const float* __restrict__ b3g,
    __bf16* sW1T, __bf16* sW2T, __bf16* sW3T,
    float* sB1, float* sB2, float* sB3) {
  const size_t o1 = ((size_t)i * 4 + j) * 4096;
  for (int e = tid; e < 4096; e += 256) {
    int k = e >> 6, nn = e & 63;
    sW1T[nn * WK64_STRIDE + k] = (__bf16)W1g[o1 + e];
  }
  const size_t o2 = ((size_t)i * 4 + j) * 2048;
  for (int e = tid; e < 2048; e += 256) {
    int k = e >> 5, nn = e & 31;
    sW2T[nn * WK64_STRIDE + k] = (__bf16)W2g[o2 + e];
  }
  for (int e = tid; e < 2048; e += 256) {
    int k = e >> 6, nn = e & 63;
    sW3T[nn * WK32_STRIDE + k] = (__bf16)W3g[o2 + e];
  }
  if (tid < 64)       sB1[tid]       = b1g[(size_t)i * 256 + j * 64 + tid];
  else if (tid < 96)  sB2[tid - 64]  = b2g[(size_t)i * 128 + j * 32 + (tid - 64)];
  else if (tid < 160) sB3[tid - 96]  = b3g[(size_t)i * 256 + j * 64 + (tid - 96)];
}

// Prefetch the next stage's weights toward GL2 (global_prefetch_b8). 256 lines.
static __device__ __forceinline__ void prefetch_weights(
    int tid, int i, int j, const float* __restrict__ W1g,
    const float* __restrict__ W2g, const float* __restrict__ W3g) {
  const float* p1 = W1g + ((size_t)i * 4 + j) * 4096;
  const float* p2 = W2g + ((size_t)i * 4 + j) * 2048;
  const float* p3 = W3g + ((size_t)i * 4 + j) * 2048;
  if (tid < 128)      __builtin_prefetch(p1 + tid * 32, 0, 1);
  else if (tid < 192) __builtin_prefetch(p2 + (tid - 128) * 32, 0, 1);
  else                __builtin_prefetch(p3 + (tid - 192) * 32, 0, 1);
}

__global__ __launch_bounds__(256)
void flow_kernel(const float* __restrict__ xg, const float* __restrict__ pg,
                 const float* __restrict__ W1g, const float* __restrict__ b1g,
                 const float* __restrict__ W2g, const float* __restrict__ b2g,
                 const float* __restrict__ W3g, const float* __restrict__ b3g,
                 const int* __restrict__ permg, float* __restrict__ outg) {
  extern __shared__ char smem[];
  float*  sState = (float*)(smem + OFF_STATE);
  __bf16* sH2    = (__bf16*)(smem + OFF_H2);
  __bf16* sH1    = (__bf16*)(smem + OFF_H1);
  __bf16* sXin   = (__bf16*)(smem + OFF_XIN);
  __bf16* sW1T   = (__bf16*)(smem + OFF_W1T);
  __bf16* sW2T   = (__bf16*)(smem + OFF_W2T);
  __bf16* sW3T   = (__bf16*)(smem + OFF_W3T);
  float*  sB1    = (float*)(smem + OFF_B1);
  float*  sB2    = (float*)(smem + OFF_B2);
  float*  sB3    = (float*)(smem + OFF_B3);
  int*    sPerm  = (int*)(smem + OFF_PERM);

  const int tid     = threadIdx.x;
  const int lane    = tid & 31;
  const int wave    = tid >> 5;
  const int rowBase = wave * 16;
  const size_t wgRow0 = (size_t)blockIdx.x * 128;

  // ---- load x and 0.01*pad_noise -> fp32 LDS state (128 rows x 128 cols) ----
  for (int q = tid; q < 128 * 16; q += 256) {
    int row = q >> 4, c4 = q & 15;
    float4 v = ((const float4*)(xg + (wgRow0 + row) * 64))[c4];
    ((float4*)(sState + row * ST_STRIDE))[c4] = v;
    float4 w = ((const float4*)(pg + (wgRow0 + row) * 64))[c4];
    w.x *= 0.01f; w.y *= 0.01f; w.z *= 0.01f; w.w *= 0.01f;
    ((float4*)(sState + row * ST_STRIDE + 64))[c4] = w;
  }

  float*  stRow  = sState + rowBase * ST_STRIDE;
  __bf16* h2Row  = sH2    + rowBase * A64_STRIDE;
  __bf16* h1Row  = sH1    + rowBase * A32_STRIDE;
  __bf16* xinRow = sXin   + rowBase * A64_STRIDE;

#pragma unroll 1
  for (int i = 0; i < NBLOCKS; ++i) {
    // ---- bf16 mirror of x2 (state cols 64..127); wave-local, no sync needed ----
#pragma unroll 1
    for (int r = 0; r < 16; ++r) {
      float2 v = *(const float2*)(stRow + r * ST_STRIDE + 64 + lane * 2);
      ((unsigned*)(xinRow + r * A64_STRIDE))[lane] = pack_bf16x2(v.x, v.y);
    }

    FragF tacc[4], sacc[4];

    // ---- stage j=3 (t2) ; compute ----
    __syncthreads();
    stage_weights(tid, i, 3, W1g, W2g, W3g, b1g, b2g, b3g,
                  sW1T, sW2T, sW3T, sB1, sB2, sB3);
    if (tid >= 96 && tid < 224) sPerm[tid - 96] = permg[i * 128 + (tid - 96)];
    __syncthreads();
    prefetch_weights(tid, i, 2, W1g, W2g, W3g);
    mlp_forward(xinRow, h2Row, h1Row, sW1T, sW2T, sW3T, sB1, sB2, sB3, lane, tacc);

    // ---- stage j=2 (s2) ; compute ; y1 epilogue ----
    __syncthreads();
    stage_weights(tid, i, 2, W1g, W2g, W3g, b1g, b2g, b3g,
                  sW1T, sW2T, sW3T, sB1, sB2, sB3);
    __syncthreads();
    prefetch_weights(tid, i, 1, W1g, W2g, W3g);
    mlp_forward(xinRow, h2Row, h1Row, sW1T, sW2T, sW3T, sB1, sB2, sB3, lane, sacc);
    // y1 = exp(1.272*atan(s2))*x1 + t2 -> state cols 0..63 + bf16 mirror
    couple_epilogue(stRow, xinRow, sacc, tacc, lane, 0, 0, true);

    // ---- stage j=1 (t1) ; compute ----
    __syncthreads();
    stage_weights(tid, i, 1, W1g, W2g, W3g, b1g, b2g, b3g,
                  sW1T, sW2T, sW3T, sB1, sB2, sB3);
    __syncthreads();
    prefetch_weights(tid, i, 0, W1g, W2g, W3g);
    mlp_forward(xinRow, h2Row, h1Row, sW1T, sW2T, sW3T, sB1, sB2, sB3, lane, tacc);

    // ---- stage j=0 (s1) ; compute ; y2 epilogue ----
    __syncthreads();
    stage_weights(tid, i, 0, W1g, W2g, W3g, b1g, b2g, b3g,
                  sW1T, sW2T, sW3T, sB1, sB2, sB3);
    __syncthreads();
    if (i + 1 < NBLOCKS) prefetch_weights(tid, i + 1, 3, W1g, W2g, W3g);
    mlp_forward(xinRow, h2Row, h1Row, sW1T, sW2T, sW3T, sB1, sB2, sB3, lane, sacc);
    // y2 = exp(1.272*atan(s1))*x2 + t1 -> state cols 64..127 (no mirror needed)
    couple_epilogue(stRow, xinRow, sacc, tacc, lane, 64, 64, false);

    // ---- permutation: newh[:,c] = h[:,perm[c]] ; wave owns its 16 rows ----
#pragma unroll 1
    for (int r = 0; r < 16; ++r) {
      float* row = stRow + r * ST_STRIDE;
      float v0 = row[sPerm[lane]];
      float v1 = row[sPerm[lane + 32]];
      float v2 = row[sPerm[lane + 64]];
      float v3 = row[sPerm[lane + 96]];
      row[lane]      = v0;
      row[lane + 32] = v1;
      row[lane + 64] = v2;
      row[lane + 96] = v3;
    }
  }

  // ---- store final 128-wide rows ----
#pragma unroll 1
  for (int r = 0; r < 16; ++r) {
    int row = rowBase + r;
    ((float4*)(outg + (wgRow0 + row) * 128))[lane] =
        ((const float4*)(sState + row * ST_STRIDE))[lane];
  }
}

extern "C" void kernel_launch(void* const* d_in, const int* in_sizes, int n_in,
                              void* d_out, int out_size, void* d_ws, size_t ws_size,
                              hipStream_t stream) {
  const float* xg    = (const float*)d_in[0];
  const float* pg    = (const float*)d_in[1];
  const float* W1g   = (const float*)d_in[2];
  const float* b1g   = (const float*)d_in[3];
  const float* W2g   = (const float*)d_in[4];
  const float* b2g   = (const float*)d_in[5];
  const float* W3g   = (const float*)d_in[6];
  const float* b3g   = (const float*)d_in[7];
  const int*   permg = (const int*)d_in[8];
  float* outg = (float*)d_out;

  const int B = in_sizes[0] / 64;      // 262144
  const int grid = B / 128;            // 2048 workgroups of 128 rows

  (void)d_ws; (void)ws_size; (void)n_in; (void)out_size;
  hipFuncSetAttribute((const void*)flow_kernel,
                      hipFuncAttributeMaxDynamicSharedMemorySize, SMEM_BYTES);
  flow_kernel<<<grid, 256, SMEM_BYTES, stream>>>(xg, pg, W1g, b1g, W2g, b2g,
                                                 W3g, b3g, permg, outg);
}
